// PENNLayer_24721831756521
// MI455X (gfx1250) — matrix operand order repaired and checked by reading, not compiled, not run on hardware
//
#include <hip/hip_runtime.h>

#define NN 50000
#define EE 640000
#define DD 128
#define EDIM 64
#define HH 128
#define EPSF 1e-5f

typedef unsigned short u16;
typedef unsigned int   u32;
typedef __bf16 v16bf __attribute__((ext_vector_type(16)));
typedef float  v8f   __attribute__((ext_vector_type(8)));

union AB16 { uint4 q[2]; u16 u[16]; v16bf v; };
union PK8 { u16 r[8]; uint4 q; };
union PK4 { u16 r[4]; uint2 q; };

__device__ __forceinline__ u16 f2bf(float f) {
    u32 u = __float_as_uint(f);
    u += 0x7FFFu + ((u >> 16) & 1u);
    return (u16)(u >> 16);
}
__device__ __forceinline__ float bf2f(u16 b) {
    return __uint_as_float(((u32)b) << 16);
}

// ---------------- init / packing kernels ----------------

__global__ void k_zero(float* agg, float* st1, float* st2) {
    long i = (long)blockIdx.x * blockDim.x + threadIdx.x;
    long tot4 = (long)NN * HH / 4;
    if (i < tot4) ((float4*)agg)[i] = make_float4(0.f, 0.f, 0.f, 0.f);
    if (blockIdx.x == 0 && threadIdx.x < 256) {
        st1[threadIdx.x] = 0.f;
        st2[threadIdx.x] = 0.f;
    }
}

__global__ void k_pack_h(const float* __restrict__ h, u16* __restrict__ hbf) {
    long i = (long)blockIdx.x * blockDim.x + threadIdx.x; // one per 8 elems
    long tot = (long)NN * DD / 8;
    if (i >= tot) return;
    const float4* s = (const float4*)h + i * 2;
    float4 a = s[0], b = s[1];
    PK8 p;
    p.r[0] = f2bf(a.x); p.r[1] = f2bf(a.y); p.r[2] = f2bf(a.z); p.r[3] = f2bf(a.w);
    p.r[4] = f2bf(b.x); p.r[5] = f2bf(b.y); p.r[6] = f2bf(b.z); p.r[7] = f2bf(b.w);
    *(uint4*)(hbf + i * 8) = p.q;
}

// Pack a [KT*32 x 128] fp32 weight into bf16 WMMA-B lane layout:
// packed[((kt*8+nt)*32 + lane)*16 + i] = W[(kt*32 + (i&7) + 8*(lane>>4) + 16*(i>>3))*128 + nt*16 + (lane&15)]
__global__ void k_pack_w(const float* __restrict__ W, u16* __restrict__ Wp, int KT) {
    int i = blockIdx.x * blockDim.x + threadIdx.x;
    int tot = KT * 8 * 32 * 16;
    if (i >= tot) return;
    int e  = i & 15;
    int l  = (i >> 4) & 31;
    int nt = (i >> 9) & 7;
    int kt = i >> 12;
    int k = kt * 32 + (e & 7) + 8 * (l >> 4) + 16 * (e >> 3);
    int n = nt * 16 + (l & 15);
    Wp[i] = f2bf(W[(long)k * HH + n]);
}

// ---------------- WMMA wave helpers ----------------

// Each wave computes a 16(M) x 64(N) tile: 4 accumulators, K = KT*32.
template <int KT>
__device__ __forceinline__ void wave_gemm(const u16* sA, int rstride,
                                          const u16* __restrict__ Wp, v8f acc[4],
                                          int lane, int mt, int ntBase) {
    const int m = lane & 15, g = lane >> 4;
#pragma unroll
    for (int kt = 0; kt < KT; ++kt) {
        AB16 A;
        const u16* ap = sA + (mt * 16 + m) * rstride + kt * 32 + 8 * g;
        A.q[0] = *(const uint4*)(ap);
        A.q[1] = *(const uint4*)(ap + 16);
#pragma unroll
        for (int j = 0; j < 4; ++j) {
            AB16 B;
            const uint4* bp = (const uint4*)(Wp + (size_t)((kt * 8 + ntBase + j) * 32 + lane) * 16);
            B.q[0] = bp[0];
            B.q[1] = bp[1];
            acc[j] = __builtin_amdgcn_wmma_f32_16x16x32_bf16(
                false, A.v, false, B.v, (short)0, acc[j], false, false);
        }
    }
}

// Same, but applies per-K-column affine + ReLU to the A fragment (fused BN apply).
__device__ __forceinline__ void wave_gemm_affine(const u16* sA, int rstride,
                                                 const u16* __restrict__ Wp, v8f acc[4],
                                                 int lane, int mt, int ntBase,
                                                 const float* __restrict__ av,
                                                 const float* __restrict__ cv) {
    const int m = lane & 15, g = lane >> 4;
#pragma unroll
    for (int kt = 0; kt < 4; ++kt) {
        const u16* ap = sA + (mt * 16 + m) * rstride + kt * 32 + 8 * g;
        uint4 lo = *(const uint4*)(ap);
        uint4 hi = *(const uint4*)(ap + 16);
        const u16* pl = (const u16*)&lo;
        const u16* ph = (const u16*)&hi;
        int k0 = kt * 32 + 8 * g;
        AB16 A;
#pragma unroll
        for (int t = 0; t < 8; ++t) {
            A.u[t]     = f2bf(fmaxf(0.f, bf2f(pl[t]) * av[k0 + t]      + cv[k0 + t]));
            A.u[8 + t] = f2bf(fmaxf(0.f, bf2f(ph[t]) * av[k0 + 16 + t] + cv[k0 + 16 + t]));
        }
#pragma unroll
        for (int j = 0; j < 4; ++j) {
            AB16 B;
            const uint4* bp = (const uint4*)(Wp + (size_t)((kt * 8 + ntBase + j) * 32 + lane) * 16);
            B.q[0] = bp[0];
            B.q[1] = bp[1];
            acc[j] = __builtin_amdgcn_wmma_f32_16x16x32_bf16(
                false, A.v, false, B.v, (short)0, acc[j], false, false);
        }
    }
}

// ---------------- edge kernels (template: stats pass vs compute pass) ----------------

template <bool STATS>
__global__ __launch_bounds__(256, 2) void k_edge(
    const float* __restrict__ ea, const int* __restrict__ ei,
    const float* __restrict__ b1, const u16* __restrict__ hbf,
    const u16* __restrict__ W1p, const u16* __restrict__ W2p,
    const float* __restrict__ b2, float* st1, float* agg) {
    __shared__ __align__(16) u16 smem[64 * 328]; // msg tile / reused as Y tile
    __shared__ int sdst[64];
    __shared__ float ssum[128], ssq[128];
    const int tid = threadIdx.x;
    const int e0 = blockIdx.x * 64;

    if (STATS && tid < 128) { ssum[tid] = 0.f; ssq[tid] = 0.f; }

    // Stage msg_in tile [64 x 320] as bf16 (row stride 328 elems = 656B, 16B aligned, bank-clean).
    for (int idx = tid; idx < 1024; idx += 256) { // h[src] cols 0..127
        int e = idx >> 4, c4 = idx & 15;
        int s = ei[e0 + e];
        *(uint4*)&smem[e * 328 + c4 * 8] = *(const uint4*)(hbf + (long)s * 128 + c4 * 8);
    }
    for (int idx = tid; idx < 1024; idx += 256) { // h[dst] cols 128..255
        int e = idx >> 4, c4 = idx & 15;
        int d = ei[EE + e0 + e];
        if (c4 == 0) sdst[e] = d;
        *(uint4*)&smem[e * 328 + 128 + c4 * 8] = *(const uint4*)(hbf + (long)d * 128 + c4 * 8);
    }
    for (int idx = tid; idx < 1024; idx += 256) { // edge_attr cols 256..319 (fp32 -> bf16)
        int e = idx >> 4, c4 = idx & 15;
        float4 v = *(const float4*)(ea + (long)(e0 + e) * EDIM + c4 * 4);
        PK4 p;
        p.r[0] = f2bf(v.x); p.r[1] = f2bf(v.y); p.r[2] = f2bf(v.z); p.r[3] = f2bf(v.w);
        *(uint2*)&smem[e * 328 + 256 + c4 * 4] = p.q;
    }
    __syncthreads();

    const int wave = tid >> 5, lane = tid & 31;
    const int mt = wave >> 1, ntBase = (wave & 1) * 4;
    const int g = lane >> 4, col_lo = lane & 15;

    v8f acc[4] = {};
    wave_gemm<10>(smem, 328, W1p, acc, lane, mt, ntBase); // X1 = msg_in @ W1

#pragma unroll
    for (int j = 0; j < 4; ++j) {
        int col = (ntBase + j) * 16 + col_lo;
        float bb = b1[col];
#pragma unroll
        for (int t = 0; t < 8; ++t) acc[j][t] += bb;
    }

    if (STATS) {
        // Per-column sum / sumsq: LDS float atomics, then 256 global atomics per block.
#pragma unroll
        for (int j = 0; j < 4; ++j) {
            int col = (ntBase + j) * 16 + col_lo;
            float s = 0.f, s2 = 0.f;
#pragma unroll
            for (int t = 0; t < 8; ++t) { float v = acc[j][t]; s += v; s2 += v * v; }
            atomicAdd(&ssum[col], s);
            atomicAdd(&ssq[col], s2);
        }
        __syncthreads();
        if (tid < 128) atomicAdd(&st1[tid], ssum[tid]);
        else if (tid < 256) atomicAdd(&st1[tid], ssq[tid - 128]);
    } else {
        __syncthreads(); // all waves done reading msg tile; reuse LDS for Y (stride 136)
        u16* ybuf = smem;
#pragma unroll
        for (int j = 0; j < 4; ++j) {
            int col = (ntBase + j) * 16 + col_lo;
            float a = st1[256 + col], c = st1[384 + col];
#pragma unroll
            for (int t = 0; t < 8; ++t) {
                int row = mt * 16 + t + 8 * g;
                ybuf[row * 136 + col] = f2bf(fmaxf(0.f, acc[j][t] * a + c));
            }
        }
        __syncthreads();
        v8f acc2[4] = {};
        wave_gemm<4>(ybuf, 136, W2p, acc2, lane, mt, ntBase); // messages = Y @ W2
#pragma unroll
        for (int j = 0; j < 4; ++j) {
            int col = (ntBase + j) * 16 + col_lo;
            float bb = b2[col];
#pragma unroll
            for (int t = 0; t < 8; ++t) {
                int row = mt * 16 + t + 8 * g;
                atomicAdd(&agg[(long)sdst[row] * HH + col], acc2[j][t] + bb);
            }
        }
    }
}

// ---------------- BN finalize ----------------

__global__ void k_bnfin(float* st, const float* __restrict__ gamma,
                        const float* __restrict__ beta, float cnt) {
    int t = threadIdx.x;
    if (t < 128) {
        float mean = st[t] / cnt;
        float var = st[128 + t] / cnt - mean * mean;
        float a = gamma[t] * rsqrtf(var + EPSF);
        st[256 + t] = a;
        st[384 + t] = beta[t] - mean * a;
    }
}

// ---------------- node pass 1: Y = [h|agg] @ W3 + b3, stats, store Y bf16 ----------------

__global__ __launch_bounds__(256, 2) void k_node1(
    const u16* __restrict__ hbf, const float* __restrict__ agg,
    const u16* __restrict__ W3p, const float* __restrict__ b3,
    float* st2, u16* __restrict__ ybf) {
    __shared__ __align__(16) u16 smem[64 * 264];
    __shared__ float ssum[128], ssq[128];
    const int tid = threadIdx.x;
    const int n0 = blockIdx.x * 64;
    if (tid < 128) { ssum[tid] = 0.f; ssq[tid] = 0.f; }

    for (int idx = tid; idx < 1024; idx += 256) { // h cols 0..127
        int e = idx >> 4, c4 = idx & 15, r = n0 + e;
        uint4 v = (r < NN) ? *(const uint4*)(hbf + (long)r * 128 + c4 * 8)
                           : make_uint4(0u, 0u, 0u, 0u);
        *(uint4*)&smem[e * 264 + c4 * 8] = v;
    }
    for (int idx = tid; idx < 2048; idx += 256) { // agg cols 128..255 (fp32 -> bf16)
        int e = idx >> 5, c4 = idx & 31, r = n0 + e;
        PK4 p; p.r[0] = p.r[1] = p.r[2] = p.r[3] = 0;
        if (r < NN) {
            float4 v = *(const float4*)(agg + (long)r * 128 + c4 * 4);
            p.r[0] = f2bf(v.x); p.r[1] = f2bf(v.y); p.r[2] = f2bf(v.z); p.r[3] = f2bf(v.w);
        }
        *(uint2*)&smem[e * 264 + 128 + c4 * 4] = p.q;
    }
    __syncthreads();

    const int wave = tid >> 5, lane = tid & 31;
    const int mt = wave >> 1, ntBase = (wave & 1) * 4;
    const int g = lane >> 4, col_lo = lane & 15;

    v8f acc[4] = {};
    wave_gemm<8>(smem, 264, W3p, acc, lane, mt, ntBase);

#pragma unroll
    for (int j = 0; j < 4; ++j) {
        int col = (ntBase + j) * 16 + col_lo;
        float bb = b3[col];
        float s = 0.f, s2 = 0.f;
#pragma unroll
        for (int t = 0; t < 8; ++t) {
            int r = n0 + mt * 16 + t + 8 * g;
            float v = acc[j][t] + bb;
            if (r < NN) {
                s += v; s2 += v * v;
                ybf[(long)r * HH + col] = f2bf(v);
            }
        }
        atomicAdd(&ssum[col], s);
        atomicAdd(&ssq[col], s2);
    }
    __syncthreads();
    if (tid < 128) atomicAdd(&st2[tid], ssum[tid]);
    else if (tid < 256) atomicAdd(&st2[tid], ssq[tid - 128]);
}

// ---------------- node pass 2: relu(bn(Y)) @ W4 + b4, residual, LayerNorm ----------------

__global__ __launch_bounds__(256, 2) void k_node2(
    const float* __restrict__ h, const u16* __restrict__ ybf,
    const u16* __restrict__ W4p, const float* __restrict__ b4,
    const float* __restrict__ st2, const float* __restrict__ lng,
    const float* __restrict__ lnb, float* __restrict__ out) {
    __shared__ __align__(16) char smraw[64 * 328 * 2];
    const int tid = threadIdx.x;
    const int n0 = blockIdx.x * 64;

    u16* ybuf = (u16*)smraw; // stride 136
    for (int idx = tid; idx < 1024; idx += 256) {
        int e = idx >> 4, c4 = idx & 15, r = n0 + e;
        uint4 v = (r < NN) ? *(const uint4*)(ybf + (long)r * 128 + c4 * 8)
                           : make_uint4(0u, 0u, 0u, 0u);
        *(uint4*)&ybuf[e * 136 + c4 * 8] = v;
    }
    __syncthreads();

    const int wave = tid >> 5, lane = tid & 31;
    const int mt = wave >> 1, ntBase = (wave & 1) * 4;
    const int g = lane >> 4, col_lo = lane & 15;

    v8f acc[4] = {};
    wave_gemm_affine(ybuf, 136, W4p, acc, lane, mt, ntBase, st2 + 256, st2 + 384);
    __syncthreads(); // done with ybuf; reuse LDS for z (fp32, stride 132)

    float* zbuf = (float*)smraw;
#pragma unroll
    for (int j = 0; j < 4; ++j) {
        int col = (ntBase + j) * 16 + col_lo;
        float bb = b4[col];
#pragma unroll
        for (int t = 0; t < 8; ++t) {
            int row = mt * 16 + t + 8 * g;
            int r = n0 + row;
            float hv = (r < NN) ? h[(long)r * 128 + col] : 0.f;
            zbuf[row * 132 + col] = acc[j][t] + bb + hv;
        }
    }
    __syncthreads();

    // LayerNorm: 4 lanes per row, 32 cols each, combine with lane shuffles.
    int row = tid >> 2, q = tid & 3;
    float s = 0.f, s2 = 0.f;
    for (int c = q * 32; c < q * 32 + 32; ++c) {
        float v = zbuf[row * 132 + c];
        s += v; s2 += v * v;
    }
    s += __shfl_xor(s, 1);  s += __shfl_xor(s, 2);
    s2 += __shfl_xor(s2, 1); s2 += __shfl_xor(s2, 2);
    float mu = s * (1.f / 128.f);
    float var = s2 * (1.f / 128.f) - mu * mu;
    float rs = rsqrtf(var + EPSF);
    int r = n0 + row;
    if (r < NN) {
        for (int c = q * 32; c < q * 32 + 32; ++c)
            out[(long)r * 128 + c] = (zbuf[row * 132 + c] - mu) * rs * lng[c] + lnb[c];
    }
}

// ---------------- launcher ----------------

extern "C" void kernel_launch(void* const* d_in, const int* in_sizes, int n_in,
                              void* d_out, int out_size, void* d_ws, size_t ws_size,
                              hipStream_t stream) {
    const float* h   = (const float*)d_in[0];
    const float* ea  = (const float*)d_in[1];
    const float* W1  = (const float*)d_in[2];
    const float* b1  = (const float*)d_in[3];
    const float* g1  = (const float*)d_in[4];
    const float* be1 = (const float*)d_in[5];
    const float* W2  = (const float*)d_in[6];
    const float* b2  = (const float*)d_in[7];
    const float* W3  = (const float*)d_in[8];
    const float* b3  = (const float*)d_in[9];
    const float* g2  = (const float*)d_in[10];
    const float* be2 = (const float*)d_in[11];
    const float* W4  = (const float*)d_in[12];
    const float* b4  = (const float*)d_in[13];
    const float* lng = (const float*)d_in[14];
    const float* lnb = (const float*)d_in[15];
    const int*   ei  = (const int*)d_in[16];
    float* out = (float*)d_out;

    char* ws = (char*)d_ws;
    size_t o = 0;
    auto alloc = [&](size_t bytes) {
        size_t r = o;
        o += (bytes + 255) & ~(size_t)255;
        return r;
    };
    float* agg  = (float*)(ws + alloc((size_t)NN * HH * 4)); // 25.6 MB, zeroed each call
    u16*   hbf  = (u16*)  (ws + alloc((size_t)NN * DD * 2)); // 12.8 MB
    u16*   ybf  = (u16*)  (ws + alloc((size_t)NN * HH * 2)); // 12.8 MB
    u16*   W1p  = (u16*)  (ws + alloc(320 * 128 * 2));
    u16*   W2p  = (u16*)  (ws + alloc(128 * 128 * 2));
    u16*   W3p  = (u16*)  (ws + alloc(256 * 128 * 2));
    u16*   W4p  = (u16*)  (ws + alloc(128 * 128 * 2));
    float* st1  = (float*)(ws + alloc(512 * 4)); // sum | sumsq | a | c
    float* st2  = (float*)(ws + alloc(512 * 4));

    k_zero<<<(NN * HH / 4 + 255) / 256, 256, 0, stream>>>(agg, st1, st2);
    k_pack_h<<<(NN * DD / 8 + 255) / 256, 256, 0, stream>>>(h, hbf);
    k_pack_w<<<(10 * 4096 + 255) / 256, 256, 0, stream>>>(W1, W1p, 10);
    k_pack_w<<<(4 * 4096 + 255) / 256, 256, 0, stream>>>(W2, W2p, 4);
    k_pack_w<<<(8 * 4096 + 255) / 256, 256, 0, stream>>>(W3, W3p, 8);
    k_pack_w<<<(4 * 4096 + 255) / 256, 256, 0, stream>>>(W4, W4p, 4);

    k_edge<true><<<EE / 64, 256, 0, stream>>>(ea, ei, b1, hbf, W1p, W2p, b2, st1, agg);
    k_bnfin<<<1, 128, 0, stream>>>(st1, g1, be1, (float)EE);
    k_edge<false><<<EE / 64, 256, 0, stream>>>(ea, ei, b1, hbf, W1p, W2p, b2, st1, agg);

    k_node1<<<(NN + 63) / 64, 256, 0, stream>>>(hbf, agg, W3p, b3, st2, ybf);
    k_bnfin<<<1, 128, 0, stream>>>(st2, g2, be2, (float)NN);
    k_node2<<<(NN + 63) / 64, 256, 0, stream>>>(h, ybf, W4p, b4, st2, lng, lnb, out);
}
